// SelfAttention_49804440764928
// MI455X (gfx1250) — matrix operand order repaired and checked by reading, not compiled, not run on hardware
//
#include <hip/hip_runtime.h>

typedef __attribute__((ext_vector_type(16))) __bf16 v16bf;
typedef __attribute__((ext_vector_type(8)))  float  v8f;

#define T_SEQ 4096
#define C_DIM 1024
#define BATCH 4
#define NTOK  (BATCH * T_SEQ)  // 16384
#define SCALE 0.03125f         // 1/sqrt(1024)
#define NEG_INF (-1.0e30f)

union FragBF {
  v16bf v;
  unsigned short s[16];
  uint4 q[2];
};

__device__ __forceinline__ unsigned short f2bf(float f) {
  unsigned int u = __float_as_uint(f);
  u += 0x7fffu + ((u >> 16) & 1u);  // round-to-nearest-even
  return (unsigned short)(u >> 16);
}

__device__ __forceinline__ unsigned int pack2bf(float lo, float hi) {
  return (unsigned int)f2bf(lo) | ((unsigned int)f2bf(hi) << 16);
}

__device__ __forceinline__ v8f wmma_bf16(v16bf a, v16bf b, v8f c) {
  return __builtin_amdgcn_wmma_f32_16x16x32_bf16(false, a, false, b, (short)0, c,
                                                 false, false);
}

// ---- A fragment (16x32, 16-bit): lane r=lane&15 is row M; halfword chunks at
// k0 + (lane>>4)*8 and +16 (ISA 16-bit A layout).
__device__ __forceinline__ v16bf load_a_bf16(const unsigned short* base, int ld,
                                             int row0, int k0, int lane) {
  const int r  = lane & 15;
  const int kb = (lane >> 4) * 8;
  const unsigned short* p = base + (size_t)(row0 + r) * ld + k0 + kb;
  FragBF f;
  f.q[0] = *(const uint4*)(p);
  f.q[1] = *(const uint4*)(p + 16);
  return f.v;
}

// ---- B fragment (32x16, 16-bit): lane n=lane&15 is column N; halfword chunks
// at k0 + (lane>>4)*8 and +16 (same k order as A).
__device__ __forceinline__ v16bf load_b_bf16(const unsigned short* base, int ld,
                                             int n0, int k0, int lane) {
  const int n = lane & 15;
  const unsigned short* p = base + (size_t)(n0 + n) * ld + k0 + (lane >> 4) * 8;
  FragBF f;
  f.q[0] = *(const uint4*)(p);
  f.q[1] = *(const uint4*)(p + 16);
  return f.v;
}

// =====================================================================
// Kernel 0: one-shot fp32 -> bf16 conversion (memory-bound, runs once).
// 8 elements per thread: 2x b128 loads -> 1x b128 store of packed bf16.
// =====================================================================
__global__ __launch_bounds__(256) void cvt_f32_bf16_kernel(
    const float* __restrict__ src, unsigned short* __restrict__ dst, int n) {
  const int i = (blockIdx.x * 256 + threadIdx.x) * 8;
  if (i + 8 > n) return;
  float4 a = *(const float4*)(src + i);
  float4 b = *(const float4*)(src + i + 4);
  uint4 o;
  o.x = pack2bf(a.x, a.y);
  o.y = pack2bf(a.z, a.w);
  o.z = pack2bf(b.x, b.y);
  o.w = pack2bf(b.z, b.w);
  *(uint4*)(dst + i) = o;
}

// =====================================================================
// Kernel 1: fused QKV projection GEMM (all-bf16 operands, f32 accum).
// Out[r,d] = sum_c x[r,c]*W[d,c] + b[d]
// z=0 -> Q (bf16 row-major [NTOK,C]); z=1 -> K (same); z=2 -> V stored
// TRANSPOSED per batch: Vt[b][d][t] so attention B-frag loads are contiguous.
// WG: 256 thr / 8 waves; WG tile 64 rows x 128 cols; wave tile 16x64.
// Hot loop is pure global_load_b128 + v_wmma (no conversion VALU).
// =====================================================================
__global__ __launch_bounds__(256) void qkv_proj_kernel(
    const unsigned short* __restrict__ xb,
    const unsigned short* __restrict__ Wqb, const float* __restrict__ bq,
    const unsigned short* __restrict__ Wkb, const float* __restrict__ bk,
    const unsigned short* __restrict__ Wvb, const float* __restrict__ bv,
    unsigned short* __restrict__ Qb, unsigned short* __restrict__ Kb,
    unsigned short* __restrict__ Vt) {
  const int z = blockIdx.z;
  const unsigned short* W = (z == 0) ? Wqb : ((z == 1) ? Wkb : Wvb);
  const float* bias       = (z == 0) ? bq  : ((z == 1) ? bk  : bv);

  const int wave = threadIdx.x >> 5;
  const int lane = threadIdx.x & 31;
  const int row0 = blockIdx.x * 64 + (wave >> 1) * 16;
  const int nb   = blockIdx.y * 128 + (wave & 1) * 64;

  v8f acc[4];
#pragma unroll
  for (int j = 0; j < 4; ++j) acc[j] = (v8f){0, 0, 0, 0, 0, 0, 0, 0};

  for (int c0 = 0; c0 < C_DIM; c0 += 32) {
    v16bf a = load_a_bf16(xb, C_DIM, row0, c0, lane);
#pragma unroll
    for (int j = 0; j < 4; ++j) {
      v16bf b = load_b_bf16(W, C_DIM, nb + j * 16, c0, lane);
      acc[j] = wmma_bf16(a, b, acc[j]);
    }
  }

  const int half = lane >> 4;
  const int n    = lane & 15;
#pragma unroll
  for (int j = 0; j < 4; ++j) {
    const int col = nb + j * 16 + n;
    const float bi = bias[col];
#pragma unroll
    for (int i = 0; i < 8; ++i) {
      const int row = row0 + i + 8 * half;  // token index in [0,NTOK)
      const unsigned short h = f2bf(acc[j][i] + bi);
      if (z == 0) {
        Qb[(size_t)row * C_DIM + col] = h;
      } else if (z == 1) {
        Kb[(size_t)row * C_DIM + col] = h;
      } else {
        const int b = row >> 12;  // row / 4096
        const int t = row & (T_SEQ - 1);
        Vt[((size_t)b * C_DIM + col) * T_SEQ + t] = h;
      }
    }
  }
}

// =====================================================================
// Kernel 2: causal flash attention.
// One WG per (batch, 16-query block); 8 waves each own 128 cols of d.
// Key loop in blocks of 32; S reduced across waves in LDS; online softmax
// by lanes 0-15; P (bf16) broadcast via LDS; O accumulated in registers.
// =====================================================================
__global__ __launch_bounds__(256) void flash_attn_kernel(
    const unsigned short* __restrict__ Qb, const unsigned short* __restrict__ Kb,
    const unsigned short* __restrict__ Vt, float* __restrict__ out) {
  const int b    = blockIdx.y;
  const int q0   = blockIdx.x * 16;
  const int wave = threadIdx.x >> 5;
  const int lane = threadIdx.x & 31;
  const int half = lane >> 4;
  const int nidx = lane & 15;
  const int dbase = wave * 128;

  const unsigned short* Q = Qb + (size_t)b * T_SEQ * C_DIM;
  const unsigned short* K = Kb + (size_t)b * T_SEQ * C_DIM;
  const unsigned short* V = Vt + (size_t)b * C_DIM * T_SEQ;

  __shared__ __align__(16) float Sp[8][16][32];        // per-wave partial S
  __shared__ __align__(16) unsigned short Pb[16][32];  // probabilities (bf16)
  __shared__ float rs[16];                             // per-row rescale
  __shared__ float lrow[16];                           // final denominators

  // Q fragment pinned in registers for the whole kernel: 16 rows x 128 cols.
  v16bf qf[4];
#pragma unroll
  for (int kc = 0; kc < 4; ++kc)
    qf[kc] = load_a_bf16(Q, C_DIM, q0, dbase + kc * 32, lane);

  v8f o[8];
#pragma unroll
  for (int t = 0; t < 8; ++t) o[t] = (v8f){0, 0, 0, 0, 0, 0, 0, 0};

  float m_r = NEG_INF;  // online-softmax state, live in lanes 0-15 of wave 0
  float l_r = 0.0f;

  const int nkb = (q0 + 47) >> 5;  // key blocks needed for keys 0..q0+15
  for (int kb = 0; kb < nkb; ++kb) {
    const int s0 = kb * 32;

    if (kb + 1 < nkb)  // hint next K block toward L2/WGP$
      __builtin_prefetch(K + (size_t)(s0 + 32 + nidx) * C_DIM + dbase, 0, 0);

    // ---- partial S = Q[:,dslice] @ K[s0:s0+32, dslice]^T  (16x32, f32)
    v8f st0 = (v8f){0, 0, 0, 0, 0, 0, 0, 0};
    v8f st1 = (v8f){0, 0, 0, 0, 0, 0, 0, 0};
#pragma unroll
    for (int kc = 0; kc < 4; ++kc) {
      st0 = wmma_bf16(qf[kc], load_b_bf16(K, C_DIM, s0,      dbase + kc * 32, lane), st0);
      st1 = wmma_bf16(qf[kc], load_b_bf16(K, C_DIM, s0 + 16, dbase + kc * 32, lane), st1);
    }
#pragma unroll
    for (int i = 0; i < 8; ++i) {
      const int r = i + 8 * half;
      Sp[wave][r][nidx]      = st0[i];
      Sp[wave][r][nidx + 16] = st1[i];
    }
    __syncthreads();

    // ---- online softmax for this block (one lane per query row)
    if (threadIdx.x < 16) {
      const int r = threadIdx.x;
      const int qrow = q0 + r;
      float sv[32];
      float mx = m_r;
#pragma unroll
      for (int k = 0; k < 32; ++k) {
        float s = 0.0f;
#pragma unroll
        for (int w = 0; w < 8; ++w) s += Sp[w][r][k];
        s *= SCALE;
        if (s0 + k > qrow) s = NEG_INF;  // causal mask
        sv[k] = s;
        mx = fmaxf(mx, s);
      }
      const float corr = __expf(m_r - mx);
      float sum = 0.0f;
#pragma unroll
      for (int k = 0; k < 32; ++k) {
        const float p = __expf(sv[k] - mx);
        sum += p;
        Pb[r][k] = f2bf(p);
      }
      l_r = l_r * corr + sum;
      m_r = mx;
      rs[r] = corr;
    }
    __syncthreads();

    // ---- rescale O and accumulate O += P @ V[s0:s0+32, dslice]
#pragma unroll
    for (int i = 0; i < 8; ++i) {
      const float f = rs[i + 8 * half];
#pragma unroll
      for (int t = 0; t < 8; ++t) o[t][i] *= f;
    }
    const v16bf pf = load_a_bf16(&Pb[0][0], 32, 0, 0, lane);
#pragma unroll
    for (int t = 0; t < 8; ++t) {
      v16bf vf = load_b_bf16(V, T_SEQ, dbase + t * 16, s0, lane);
      o[t] = wmma_bf16(pf, vf, o[t]);
    }
    __syncthreads();  // protect Sp/Pb/rs for next iteration
  }

  if (threadIdx.x < 16) lrow[threadIdx.x] = l_r;
  __syncthreads();

#pragma unroll
  for (int i = 0; i < 8; ++i) {
    const int row = q0 + i + 8 * half;
    const float inv = 1.0f / lrow[i + 8 * half];
    float* op = out + ((size_t)b * T_SEQ + row) * C_DIM;
#pragma unroll
    for (int t = 0; t < 8; ++t) op[dbase + t * 16 + nidx] = o[t][i] * inv;
  }
}

extern "C" void kernel_launch(void* const* d_in, const int* in_sizes, int n_in,
                              void* d_out, int out_size, void* d_ws, size_t ws_size,
                              hipStream_t stream) {
  const float* x  = (const float*)d_in[0];
  const float* Wq = (const float*)d_in[1];
  const float* bq = (const float*)d_in[2];
  const float* Wk = (const float*)d_in[3];
  const float* bk = (const float*)d_in[4];
  const float* Wv = (const float*)d_in[5];
  const float* bv = (const float*)d_in[6];

  const size_t NTC = (size_t)NTOK * C_DIM;  // 16,777,216
  const size_t WSZ = (size_t)C_DIM * C_DIM; //  1,048,576

  unsigned short* Qb  = (unsigned short*)d_ws;  // [NTOK, C] bf16
  unsigned short* Kb  = Qb + NTC;               // [NTOK, C] bf16
  unsigned short* Vt  = Kb + NTC;               // [B, C, T] bf16
  unsigned short* xb  = Vt + NTC;               // [NTOK, C] bf16
  unsigned short* Wqb = xb + NTC;               // [C, C] bf16
  unsigned short* Wkb = Wqb + WSZ;
  unsigned short* Wvb = Wkb + WSZ;

  // One-shot conversions (memory-bound).
  cvt_f32_bf16_kernel<<<dim3((unsigned)(NTC / (256 * 8))), 256, 0, stream>>>(x, xb, (int)NTC);
  cvt_f32_bf16_kernel<<<dim3((unsigned)(WSZ / (256 * 8))), 256, 0, stream>>>(Wq, Wqb, (int)WSZ);
  cvt_f32_bf16_kernel<<<dim3((unsigned)(WSZ / (256 * 8))), 256, 0, stream>>>(Wk, Wkb, (int)WSZ);
  cvt_f32_bf16_kernel<<<dim3((unsigned)(WSZ / (256 * 8))), 256, 0, stream>>>(Wv, Wvb, (int)WSZ);

  dim3 g1(NTOK / 64, C_DIM / 128, 3);
  qkv_proj_kernel<<<g1, 256, 0, stream>>>(xb, Wqb, bq, Wkb, bk, Wvb, bv, Qb, Kb, Vt);

  dim3 g2(T_SEQ / 16, BATCH);
  flash_attn_kernel<<<g2, 256, 0, stream>>>(Qb, Kb, Vt, (float*)d_out);
}
